// PointNetSetAbstraction_42597485641816
// MI455X (gfx1250) — matrix operand order repaired and checked by reading, not compiled, not run on hardware
//
#include <hip/hip_runtime.h>
#include <hip/hip_bf16.h>
#include <stdint.h>

// Problem constants (from the reference)
#define Bb   16
#define Nn   4096
#define Ss   1024
#define Kk   32
#define CPTS 64
#define RR   (Bb * Ss * Kk)   // 524288 rows through the MLP

typedef __attribute__((ext_vector_type(16))) __bf16 v16bf;
typedef __attribute__((ext_vector_type(8)))  float  v8f;
typedef __attribute__((ext_vector_type(4)))  unsigned int u32x4;

__device__ __forceinline__ unsigned short f2bf(float f) {
    unsigned int u = __float_as_uint(f);
    u += 0x7FFFu + ((u >> 16) & 1u);          // round-to-nearest-even
    return (unsigned short)(u >> 16);
}
__device__ __forceinline__ float bf2f(unsigned short h) {
    return __uint_as_float(((unsigned int)h) << 16);
}

// ---------------------------------------------------------------------------
// 1) Farthest point sampling: one block per batch, 256 threads, points cached
//    in registers (16/thread). Sequential 1024-step loop with LDS argmax.
// ---------------------------------------------------------------------------
__global__ __launch_bounds__(256) void fps_kernel(const float* __restrict__ xyz,
                                                  int* __restrict__ fps_idx,
                                                  float* __restrict__ new_xyz) {
    const int b   = blockIdx.x;
    const int tid = threadIdx.x;
    const float* px = xyz + (size_t)b * Nn * 3;

    float lx[16], ly[16], lz[16], dmin[16];
#pragma unroll
    for (int j = 0; j < 16; ++j) {
        int p = tid * 16 + j;
        lx[j] = px[p * 3 + 0];
        ly[j] = px[p * 3 + 1];
        lz[j] = px[p * 3 + 2];
        dmin[j] = 1e10f;
    }

    __shared__ float rv[256];
    __shared__ int   ri[256];
    __shared__ int   sFar;
    if (tid == 0) sFar = 0;
    __syncthreads();

    for (int s = 0; s < Ss; ++s) {
        const int far = sFar;
        if (tid == 0) {
            fps_idx[b * Ss + s] = far;
            new_xyz[(b * Ss + s) * 3 + 0] = px[far * 3 + 0];
            new_xyz[(b * Ss + s) * 3 + 1] = px[far * 3 + 1];
            new_xyz[(b * Ss + s) * 3 + 2] = px[far * 3 + 2];
        }
        const float cx = px[far * 3 + 0];
        const float cy = px[far * 3 + 1];
        const float cz = px[far * 3 + 2];

        float bv = -1.0f; int bi = 0;
#pragma unroll
        for (int j = 0; j < 16; ++j) {
            float dx = lx[j] - cx, dy = ly[j] - cy, dz = lz[j] - cz;
            float d  = dx * dx + dy * dy + dz * dz;
            float dm = fminf(dmin[j], d);
            dmin[j] = dm;
            if (dm > bv) { bv = dm; bi = tid * 16 + j; }
        }
        rv[tid] = bv; ri[tid] = bi;
        __syncthreads();
        for (int off = 128; off > 0; off >>= 1) {
            if (tid < off) {
                float ov = rv[tid + off]; int oi = ri[tid + off];
                if (ov > rv[tid] || (ov == rv[tid] && oi < ri[tid])) {
                    rv[tid] = ov; ri[tid] = oi;
                }
            }
            __syncthreads();
        }
        if (tid == 0) sFar = ri[0];
        __syncthreads();
    }
}

// ---------------------------------------------------------------------------
// 2) KNN: one wave32 per centroid. Distances into a private LDS slab, then
//    32 x extract-min with shfl_xor butterfly (wave32) selection.
// ---------------------------------------------------------------------------
__global__ __launch_bounds__(64) void knn_kernel(const float* __restrict__ xyz,
                                                 const float* __restrict__ new_xyz,
                                                 int* __restrict__ knn_idx) {
    __shared__ float dist[2][Nn];     // 32 KB
    const int lane = threadIdx.x & 31;
    const int wid  = threadIdx.x >> 5;
    const int task = blockIdx.x * 2 + wid;          // b*S + s
    const int b    = task / Ss;

    const float cx = new_xyz[task * 3 + 0];
    const float cy = new_xyz[task * 3 + 1];
    const float cz = new_xyz[task * 3 + 2];
    const float* px = xyz + (size_t)b * Nn * 3;

    for (int j = lane; j < Nn; j += 32) {
        float dx = px[j * 3 + 0] - cx;
        float dy = px[j * 3 + 1] - cy;
        float dz = px[j * 3 + 2] - cz;
        dist[wid][j] = dx * dx + dy * dy + dz * dz;
    }
    __syncthreads();

    for (int it = 0; it < Kk; ++it) {
        float bv = 3.0e38f; int bi = 0;
        for (int j = lane; j < Nn; j += 32) {
            float v = dist[wid][j];
            if (v < bv || (v == bv && j < bi)) { bv = v; bi = j; }
        }
        for (int off = 16; off > 0; off >>= 1) {
            float ov = __shfl_xor(bv, off, 32);
            int   oi = __shfl_xor(bi, off, 32);
            if (ov < bv || (ov == bv && oi < bi)) { bv = ov; bi = oi; }
        }
        if (lane == 0) {
            knn_idx[task * Kk + it] = bi;
            dist[wid][bi] = 3.0e38f;
        }
        __syncthreads();
    }
}

// ---------------------------------------------------------------------------
// 3) Build X0 [R, 96] bf16 : [centered xyz (3) | grouped points (64) | pad 0]
//    One wave per row, lanes cover channels.
// ---------------------------------------------------------------------------
__global__ __launch_bounds__(256) void group_kernel(const float* __restrict__ xyz,
                                                    const float* __restrict__ points,
                                                    const float* __restrict__ new_xyz,
                                                    const int* __restrict__ knn_idx,
                                                    unsigned short* __restrict__ X0) {
    const int lane = threadIdx.x & 31;
    const int wid  = threadIdx.x >> 5;
    const int r    = blockIdx.x * 8 + wid;          // row = (b*S+s)*K + k
    const int bs   = r >> 5;                        // b*S + s
    const int b    = r >> 15;
    const int idx  = knn_idx[r];

    const float* xsrc = xyz    + ((size_t)b * Nn + idx) * 3;
    const float* psrc = points + ((size_t)b * Nn + idx) * CPTS;
    unsigned short* dst = X0 + (size_t)r * 96;

#pragma unroll
    for (int t = 0; t < 3; ++t) {
        int c = lane + t * 32;
        float v;
        if (c < 3)       v = xsrc[c] - new_xyz[bs * 3 + c];
        else if (c < 67) v = psrc[c - 3];
        else             v = 0.0f;
        dst[c] = f2bf(v);
    }
}

// ---------------------------------------------------------------------------
// Utility: zero the BN stats accumulators
// ---------------------------------------------------------------------------
__global__ void zero_kernel(float* __restrict__ p, int n) {
    int i = blockIdx.x * 256 + threadIdx.x;
    if (i < n) p[i] = 0.0f;
}

// ---------------------------------------------------------------------------
// 4) Fused 1x1-conv layer: Y = relu_bn_in(X) @ W + bias, via bf16 WMMA.
//    8 waves/block, one 16-row M-tile per wave, NT column tiles of 16.
//    Accumulates per-channel sum / sum^2 (BN train stats) via LDS + atomics.
// ---------------------------------------------------------------------------
template <int NT, int KC>
__global__ __launch_bounds__(256) void mlp_matmul(
        const unsigned short* __restrict__ X, int ldx, int cin,
        const float* __restrict__ W, const float* __restrict__ bias,
        const float* __restrict__ inScale, const float* __restrict__ inShift,
        int applyInBN,
        unsigned short* __restrict__ Y, int ldy, int storeY,
        float* __restrict__ gSum, float* __restrict__ gSq) {
    constexpr int KP   = KC * 32;
    constexpr int COUT = NT * 16;
    __shared__ __align__(16) unsigned short wlds[COUT * KP];   // transposed [n][k]
    __shared__ float ldsSum[COUT], ldsSq[COUT];

    const int tid = threadIdx.x;
    for (int i = tid; i < COUT; i += 256) { ldsSum[i] = 0.f; ldsSq[i] = 0.f; }
    for (int i = tid; i < COUT * KP; i += 256) {
        int n = i / KP, k = i % KP;
        wlds[i] = (k < cin) ? f2bf(W[(size_t)k * COUT + n]) : (unsigned short)0;
    }
    __syncthreads();

    const int lane = tid & 31;
    const int wid  = tid >> 5;
    const int half = lane >> 4;
    const int m    = lane & 15;
    const int r0   = (blockIdx.x * 8 + wid) * 16;

    union Frag { v16bf v; unsigned short u[16]; u32x4 q[2]; };

    v8f zero8 = {0.f, 0.f, 0.f, 0.f, 0.f, 0.f, 0.f, 0.f};
    v8f acc[NT];
#pragma unroll
    for (int nt = 0; nt < NT; ++nt) acc[nt] = zero8;

    const unsigned short* rowp = X + (size_t)(r0 + m) * ldx;

#pragma unroll
    for (int kc = 0; kc < KC; ++kc) {
        const int k0 = kc * 32;
        const int kA = k0 + half * 8;          // K = kA..kA+7   (u[0..7])
        const int kB = k0 + 16 + half * 8;     // K = kB..kB+7   (u[8..15])
        Frag a;
        a.q[0] = *reinterpret_cast<const u32x4*>(rowp + kA);
        a.q[1] = *reinterpret_cast<const u32x4*>(rowp + kB);
        if (applyInBN) {
#pragma unroll
            for (int j = 0; j < 16; ++j) {
                int ch = (j < 8) ? (kA + j) : (kB + j - 8);
                float f = bf2f(a.u[j]);
                f = fmaxf(f * inScale[ch] + inShift[ch], 0.0f);
                a.u[j] = f2bf(f);
            }
        }
#pragma unroll
        for (int nt = 0; nt < NT; ++nt) {
            const int n = nt * 16 + m;
            Frag bfr;
            const unsigned short* wp = wlds + n * KP + k0 + half * 8;
            bfr.q[0] = *reinterpret_cast<const u32x4*>(wp);
            bfr.q[1] = *reinterpret_cast<const u32x4*>(wp + 16);
            acc[nt] = __builtin_amdgcn_wmma_f32_16x16x32_bf16(
                false, a.v, false, bfr.v, (short)0, acc[nt], false, false);
        }
    }

#pragma unroll
    for (int nt = 0; nt < NT; ++nt) {
        const int   n  = nt * 16 + m;
        const float bv = bias[n];
        float s = 0.f, q = 0.f;
        float d[8];
#pragma unroll
        for (int v = 0; v < 8; ++v) {
            d[v] = acc[nt][v] + bv;
            s += d[v];
            q += d[v] * d[v];
        }
        s += __shfl_xor(s, 16, 32);            // fold the two half-lane row sets
        q += __shfl_xor(q, 16, 32);
        if (lane < 16) {
            atomicAdd(&ldsSum[n], s);
            atomicAdd(&ldsSq[n], q);
        }
        if (storeY) {
#pragma unroll
            for (int v = 0; v < 8; ++v)
                Y[(size_t)(r0 + v + 8 * half) * ldy + n] = f2bf(d[v]);
        }
    }
    __syncthreads();
    for (int i = tid; i < COUT; i += 256) {
        atomicAdd(&gSum[i], ldsSum[i]);
        atomicAdd(&gSq[i], ldsSq[i]);
    }
}

// ---------------------------------------------------------------------------
// 5) BN finalize: scale = g * rsqrt(var+eps); shift = be - mean*scale
// ---------------------------------------------------------------------------
__global__ void bn_finalize(const float* __restrict__ gSum, const float* __restrict__ gSq,
                            const float* __restrict__ g, const float* __restrict__ be,
                            float* __restrict__ scale, float* __restrict__ shift, int cout) {
    int c = threadIdx.x;
    if (c < cout) {
        const float inv  = 1.0f / (float)RR;
        float mean = gSum[c] * inv;
        float var  = gSq[c] * inv - mean * mean;
        float sc   = g[c] * rsqrtf(var + 1e-5f);
        scale[c] = sc;
        shift[c] = be[c] - mean * sc;
    }
}

// ---------------------------------------------------------------------------
// 6) Layer-2 recompute + BN + relu + max over K=32, one wave per (b,s).
//    Two 16-row M-tiles per wave; max folded across D VGPRs and lane halves.
// ---------------------------------------------------------------------------
__global__ __launch_bounds__(256) void pool_kernel(
        const unsigned short* __restrict__ Y1,
        const float* __restrict__ W2, const float* __restrict__ b2,
        const float* __restrict__ scale1, const float* __restrict__ shift1,
        const float* __restrict__ scale2, const float* __restrict__ shift2,
        float* __restrict__ out) {
    constexpr int KP = 64, COUT = 128;
    __shared__ __align__(16) unsigned short wlds[COUT * KP];   // transposed [n][k]
    const int tid = threadIdx.x;
    for (int i = tid; i < COUT * KP; i += 256) {
        int n = i / KP, k = i % KP;
        wlds[i] = f2bf(W2[(size_t)k * COUT + n]);
    }
    __syncthreads();

    const int lane = tid & 31, wid = tid >> 5;
    const int half = lane >> 4, m = lane & 15;
    const int task = blockIdx.x * 8 + wid;        // b*S + s
    const int r0   = task * 32;

    union Frag { v16bf v; unsigned short u[16]; u32x4 q[2]; };
    Frag a[2][2];   // [mtile][kchunk], BN1+relu applied on load
#pragma unroll
    for (int mt = 0; mt < 2; ++mt) {
        const unsigned short* rowp = Y1 + (size_t)(r0 + mt * 16 + m) * 64;
#pragma unroll
        for (int kc = 0; kc < 2; ++kc) {
            int k0 = kc * 32, kA = k0 + half * 8, kB = k0 + 16 + half * 8;
            a[mt][kc].q[0] = *reinterpret_cast<const u32x4*>(rowp + kA);
            a[mt][kc].q[1] = *reinterpret_cast<const u32x4*>(rowp + kB);
#pragma unroll
            for (int j = 0; j < 16; ++j) {
                int ch = (j < 8) ? (kA + j) : (kB + j - 8);
                float f = bf2f(a[mt][kc].u[j]);
                f = fmaxf(f * scale1[ch] + shift1[ch], 0.0f);
                a[mt][kc].u[j] = f2bf(f);
            }
        }
    }

    v8f zero8 = {0.f, 0.f, 0.f, 0.f, 0.f, 0.f, 0.f, 0.f};
#pragma unroll
    for (int nt = 0; nt < 8; ++nt) {
        const int n = nt * 16 + m;
        Frag bf0, bf1;
        const unsigned short* wp = wlds + n * KP;
        bf0.q[0] = *reinterpret_cast<const u32x4*>(wp + half * 8);
        bf0.q[1] = *reinterpret_cast<const u32x4*>(wp + half * 8 + 16);
        bf1.q[0] = *reinterpret_cast<const u32x4*>(wp + 32 + half * 8);
        bf1.q[1] = *reinterpret_cast<const u32x4*>(wp + 32 + half * 8 + 16);

        v8f acc0 = zero8, acc1 = zero8;
        acc0 = __builtin_amdgcn_wmma_f32_16x16x32_bf16(false, a[0][0].v, false, bf0.v, (short)0, acc0, false, false);
        acc0 = __builtin_amdgcn_wmma_f32_16x16x32_bf16(false, a[0][1].v, false, bf1.v, (short)0, acc0, false, false);
        acc1 = __builtin_amdgcn_wmma_f32_16x16x32_bf16(false, a[1][0].v, false, bf0.v, (short)0, acc1, false, false);
        acc1 = __builtin_amdgcn_wmma_f32_16x16x32_bf16(false, a[1][1].v, false, bf1.v, (short)0, acc1, false, false);

        const float sc = scale2[n], sh = shift2[n], bb = b2[n];
        float mx = -3.0e38f;
#pragma unroll
        for (int v = 0; v < 8; ++v) {
            float y0 = fmaxf((acc0[v] + bb) * sc + sh, 0.0f);
            float y1 = fmaxf((acc1[v] + bb) * sc + sh, 0.0f);
            mx = fmaxf(mx, fmaxf(y0, y1));
        }
        mx = fmaxf(mx, __shfl_xor(mx, 16, 32));   // combine the two row halves
        if (lane < 16)
            out[(size_t)task * 128 + n] = mx;
    }
}

// ---------------------------------------------------------------------------
// Host-side orchestration
// ---------------------------------------------------------------------------
extern "C" void kernel_launch(void* const* d_in, const int* in_sizes, int n_in,
                              void* d_out, int out_size, void* d_ws, size_t ws_size,
                              hipStream_t stream) {
    const float* xyz    = (const float*)d_in[0];
    const float* points = (const float*)d_in[1];
    const float* w0 = (const float*)d_in[2];
    const float* b0 = (const float*)d_in[3];
    const float* g0 = (const float*)d_in[4];
    const float* e0 = (const float*)d_in[5];
    const float* w1 = (const float*)d_in[6];
    const float* b1 = (const float*)d_in[7];
    const float* g1 = (const float*)d_in[8];
    const float* e1 = (const float*)d_in[9];
    const float* w2 = (const float*)d_in[10];
    const float* b2 = (const float*)d_in[11];
    const float* g2 = (const float*)d_in[12];
    const float* e2 = (const float*)d_in[13];

    float* out        = (float*)d_out;
    float* new_xyz    = out;                       // [B,S,3]
    float* new_points = out + (size_t)Bb * Ss * 3; // [B,S,128]

    char*  ws  = (char*)d_ws;
    size_t off = 0;
    auto take = [&](size_t bytes) -> void* {
        off = (off + 255) & ~(size_t)255;
        void* p = ws + off;
        off += bytes;
        return p;
    };

    int*            fps_idx = (int*)take((size_t)Bb * Ss * 4);
    int*            knn_idx = (int*)take((size_t)RR * 4);
    unsigned short* X0      = (unsigned short*)take((size_t)RR * 96 * 2);
    unsigned short* Y0      = (unsigned short*)take((size_t)RR * 64 * 2);
    unsigned short* Y1      = (unsigned short*)take((size_t)RR * 64 * 2);
    float*          stats   = (float*)take((size_t)3 * 512 * 4); // per layer: sum|sq|scale|shift x128

    float* sum0 = stats + 0 * 512, *sq0 = sum0 + 128, *sc0 = sum0 + 256, *sh0 = sum0 + 384;
    float* sum1 = stats + 1 * 512, *sq1 = sum1 + 128, *sc1 = sum1 + 256, *sh1 = sum1 + 384;
    float* sum2 = stats + 2 * 512, *sq2 = sum2 + 128, *sc2 = sum2 + 256, *sh2 = sum2 + 384;

    zero_kernel<<<(3 * 512 + 255) / 256, 256, 0, stream>>>(stats, 3 * 512);

    fps_kernel<<<Bb, 256, 0, stream>>>(xyz, fps_idx, new_xyz);
    knn_kernel<<<Bb * Ss / 2, 64, 0, stream>>>(xyz, new_xyz, knn_idx);
    group_kernel<<<RR / 8, 256, 0, stream>>>(xyz, points, new_xyz, knn_idx, X0);

    // Layer 0: [R,67(->96)] x [67,64]
    mlp_matmul<4, 3><<<RR / 128, 256, 0, stream>>>(X0, 96, 67, w0, b0,
                                                   nullptr, nullptr, 0,
                                                   Y0, 64, 1, sum0, sq0);
    bn_finalize<<<1, 128, 0, stream>>>(sum0, sq0, g0, e0, sc0, sh0, 64);

    // Layer 1: relu(bn0(Y0)) x [64,64]
    mlp_matmul<4, 2><<<RR / 128, 256, 0, stream>>>(Y0, 64, 64, w1, b1,
                                                   sc0, sh0, 1,
                                                   Y1, 64, 1, sum1, sq1);
    bn_finalize<<<1, 128, 0, stream>>>(sum1, sq1, g1, e1, sc1, sh1, 64);

    // Layer 2 stats-only pass: relu(bn1(Y1)) x [64,128] (no Y2 store)
    mlp_matmul<8, 2><<<RR / 128, 256, 0, stream>>>(Y1, 64, 64, w2, b2,
                                                   sc1, sh1, 1,
                                                   nullptr, 0, 0, sum2, sq2);
    bn_finalize<<<1, 128, 0, stream>>>(sum2, sq2, g2, e2, sc2, sh2, 128);

    // Layer 2 recompute + bn2 + relu + max over K -> new_points
    pool_kernel<<<Bb * Ss / 8, 256, 0, stream>>>(Y1, w2, b2, sc1, sh1, sc2, sh2,
                                                 new_points);
}
